// DU_Core_V1_52003464020670
// MI455X (gfx1250) — matrix operand order repaired and checked by reading, not compiled, not run on hardware
//
#include <hip/hip_runtime.h>

// ---------------------------------------------------------------------------
// SNN (snntorch Leaky / LIF) forward pass for MI455X (gfx1250, wave32, WMMA).
// Per timestep: cur1 = x_t @ W1^T + b1 ; LIF ; cur2 = spk1 @ W2^T + b2 ; LIF.
// bf16 WMMA (v_wmma_f32_16x16x32_bf16) with f32 accumulate; weights converted
// to bf16 once (24 MB -> resident in 192 MB L2 across all 200 steps).
// Double-buffered LDS: weight tiles (and layer-2 activation tiles) are DMA'd
// by the Tensor Data Mover (tensor_load_to_lds, TENSORcnt) for iteration i+1
// while iteration i's WMMAs execute; one barrier per K-chunk.
// ---------------------------------------------------------------------------

typedef __bf16 bf16;
typedef __attribute__((ext_vector_type(16))) __bf16 v16bf;
typedef __attribute__((ext_vector_type(8)))  __bf16 v8bf;
typedef __attribute__((ext_vector_type(4)))  __bf16 v4bf;
typedef __attribute__((ext_vector_type(8)))  float  v8f;
typedef __attribute__((ext_vector_type(4)))  unsigned int v4u;
typedef __attribute__((ext_vector_type(8)))  int    v8i;
typedef __attribute__((ext_vector_type(4)))  int    v4i;

#define BATCH   128
#define IN_DIM  2048
#define HID_DIM 4096
#define OUT_DIM 1024
#define BETA    0.95f
#define THR     1.0f

// ---- workspace layout (bytes, all 256-aligned) ----------------------------
#define WS_OFF_W1B   ((size_t)0)                                   // 16 MiB
#define WS_OFF_W2B   (WS_OFF_W1B + (size_t)HID_DIM*IN_DIM*2)       //  8 MiB
#define WS_OFF_MEM1  (WS_OFF_W2B + (size_t)OUT_DIM*HID_DIM*2)      //  2 MiB
#define WS_OFF_MEM2  (WS_OFF_MEM1 + (size_t)BATCH*HID_DIM*4)       // 512 KiB
#define WS_OFF_SPK1  (WS_OFF_MEM2 + (size_t)BATCH*OUT_DIM*4)       //  1 MiB

// ---------------------------------------------------------------------------
// fp32 -> bf16 weight conversion (one-shot per launch)
// ---------------------------------------------------------------------------
__global__ __launch_bounds__(256) void f32_to_bf16_kernel(
    const float* __restrict__ src, bf16* __restrict__ dst, int n)
{
    int i = (blockIdx.x * blockDim.x + threadIdx.x) * 4;
    if (i + 3 < n) {
        float4 v = *(const float4*)(src + i);
        v4bf o;
        o[0] = (bf16)v.x; o[1] = (bf16)v.y; o[2] = (bf16)v.z; o[3] = (bf16)v.w;
        *(v4bf*)(dst + i) = o;
    }
}

// ---------------------------------------------------------------------------
// TDM: async-DMA one 2D bf16 tile [tile_d1 rows x tile_d0 elems] from global
// (row stride = stride_elems) into LDS at lds_off, inserting 4 DWORDs of LDS
// padding after every 16 DWORDs (64B row -> 80B LDS row stride).
// D# bitfields per CDNA5 ISA ch.8: group0 = {count, lds_addr, global_addr,
// type=2}; group1 = {data_size=2B, pad cfg, tensor dims, tile dims, stride}.
// ---------------------------------------------------------------------------
__device__ __forceinline__ void tdm_load_tile_bf16(
    const bf16* gptr, unsigned lds_off,
    int tensor_d0, int tensor_d1, int tile_d0, int tile_d1, int stride_elems)
{
    unsigned long long ga = (unsigned long long)(size_t)gptr;
    v4u g0;
    g0[0] = 1u;                                         // count=1 (valid D#)
    g0[1] = lds_off;                                    // LDS byte address
    g0[2] = (unsigned)(ga & 0xFFFFFFFFu);               // global_addr[31:0]
    g0[3] = (unsigned)((ga >> 32) & 0x01FFFFFFu)        // global_addr[56:32]
          | (2u << 30);                                 // type = 2 ("image")
    v8i g1;
    g1[0] = (1 << 16)                                   // data_size = 2 bytes
          | (1 << 20)                                   // pad_enable
          | (3 << 22)                                   // pad_interval: 16 DW
          | (3 << 25);                                  // pad_amount : 4 DW
    g1[1] = (tensor_d0 & 0xFFFF) << 16;                 // tensor_dim0[15:0]
    g1[2] = ((unsigned)tensor_d0 >> 16)                 // tensor_dim0[31:16]
          | ((tensor_d1 & 0xFFFF) << 16);               // tensor_dim1[15:0]
    g1[3] = ((unsigned)tensor_d1 >> 16)                 // tensor_dim1[31:16]
          | ((tile_d0 & 0xFFFF) << 16);                 // tile_dim0
    g1[4] = (tile_d1 & 0xFFFF);                         // tile_dim1 (dim2=0)
    g1[5] = stride_elems;                               // dim0_stride[31:0]
    g1[6] = 0;                                          // dim0_stride[47:32]
    g1[7] = 0;                                          // dim1_stride unused
    v4i z4 = {0, 0, 0, 0};
#if __clang_major__ >= 23
    v8i z8 = {0, 0, 0, 0, 0, 0, 0, 0};
    __builtin_amdgcn_tensor_load_to_lds(g0, g1, z4, z4, z8, 0);
#else
    __builtin_amdgcn_tensor_load_to_lds(g0, g1, z4, z4, 0);
#endif
}

// ---------------------------------------------------------------------------
// Fused GEMM + LIF step.
//   D[128 x N_TOTAL] = A[128 x K_TOTAL] * W^T,  W is [N_TOTAL x K_TOTAL] bf16.
//   mem = BETA*mem + D + bias ; spk = (mem > THR) ; mem -= spk*THR
// Block: 256 threads (8 waves), tile M=128 (full batch) x N=128, K-chunk 32.
// Wave w owns N columns [w*16, w*16+16) and 8 M-tiles of v_wmma 16x16x32 bf16.
// Double-buffered LDS; TDM DMA for chunk i+1 overlaps WMMAs of chunk i.
// ---------------------------------------------------------------------------
template<int N_TOTAL, int K_TOTAL, bool A_F32, bool OUT_BF16>
__global__ __launch_bounds__(256) void lif_gemm_kernel(
    const void* __restrict__ Asrc,        // [128][K_TOTAL] f32 or bf16
    const bf16* __restrict__ W,           // [N_TOTAL][K_TOTAL] bf16
    const float* __restrict__ bias,       // [N_TOTAL]
    float* __restrict__ mem,              // [128][N_TOTAL] persistent
    bf16*  __restrict__ spk_bf16,         // layer-1 spikes (bf16) or unused
    float* __restrict__ spk_f32)          // layer-2 spikes (f32) or unused
{
    constexpr int BK  = 32;
    constexpr int LDK = BK + 8;           // 40 elems = 80 B row stride (16B mult)
    constexpr int NK  = K_TOTAL / BK;
    __shared__ alignas(16) bf16 As[2][BATCH * LDK];
    __shared__ alignas(16) bf16 Bs[2][128   * LDK];

    const int tid    = threadIdx.x;
    const int wave   = tid >> 5;
    const int lane   = tid & 31;
    const int half   = lane >> 4;         // 0/1 : lane group
    const int l16    = lane & 15;
    const int nBlock = blockIdx.x * 128;

    // cooperative loaders (layer-1 A conversion): thread -> (row, K-half)
    const int lr  = tid >> 1;             // 0..127
    const int lkh = (tid & 1) * 16;       // 0 or 16

    v8f acc[8];
    #pragma unroll
    for (int i = 0; i < 8; ++i) acc[i] = (v8f){0,0,0,0,0,0,0,0};

    const bool issuer = (wave == 0) || (!A_F32 && wave == 1);

    // ---- stage K-chunk `kidx` into LDS buffer `buf` ----------------------
    auto stage = [&](int kidx, int buf) {
        const int k0 = kidx * BK;
        if (wave == 0) {            // W tile via TDM (HW-padded rows)
            tdm_load_tile_bf16(W + (size_t)nBlock * K_TOTAL + k0,
                               (unsigned)(size_t)(&Bs[buf][0]),
                               K_TOTAL, 128, BK, 128, K_TOTAL);
        }
        if (A_F32) {                // layer 1: VALU f32->bf16 convert+store
            bf16* adst = &As[buf][lr * LDK + lkh];
            const float* Ap = (const float*)Asrc + (size_t)lr * K_TOTAL + k0 + lkh;
            #pragma unroll
            for (int h = 0; h < 2; ++h) {
                float4 v0 = *(const float4*)(Ap + h * 8);
                float4 v1 = *(const float4*)(Ap + h * 8 + 4);
                v8bf p;
                p[0] = (bf16)v0.x; p[1] = (bf16)v0.y; p[2] = (bf16)v0.z; p[3] = (bf16)v0.w;
                p[4] = (bf16)v1.x; p[5] = (bf16)v1.y; p[6] = (bf16)v1.z; p[7] = (bf16)v1.w;
                *(v8bf*)(adst + h * 8) = p;
            }
            if (k0 + BK < K_TOTAL)
                __builtin_prefetch(Ap + BK, 0, 0);      // global_prefetch_b8
        } else if (wave == 1) {     // layer 2: A tile via a second TDM
            tdm_load_tile_bf16((const bf16*)Asrc + k0,
                               (unsigned)(size_t)(&As[buf][0]),
                               K_TOTAL, BATCH, BK, BATCH, K_TOTAL);
        }
    };

    // ---- prologue: fill buffer 0 -----------------------------------------
    stage(0, 0);
    if (issuer) __builtin_amdgcn_s_wait_tensorcnt(0);
    __syncthreads();

    // ---- main pipeline ----------------------------------------------------
    for (int it = 0; it < NK; ++it) {
        const int cur = it & 1;

        // kick off the DMA for the next chunk; it overlaps this chunk's math
        if (it + 1 < NK) stage(it + 1, (it + 1) & 1);

        // ---- B fragment: column n = wave*16 + l16, K = half*8 + {0..7, 16..23} -------
        const bf16* bp = &Bs[cur][(wave * 16 + l16) * LDK + half * 8];
        v8bf b0 = *(const v8bf*)(bp);
        v8bf b1 = *(const v8bf*)(bp + 16);
        v16bf bfrag;
        #pragma unroll
        for (int i = 0; i < 8; ++i) { bfrag[i] = b0[i]; bfrag[i + 8] = b1[i]; }

        // ---- preload all 8 A fragments (one LDS clause, one wait) --------
        v16bf afrag[8];
        #pragma unroll
        for (int mt = 0; mt < 8; ++mt) {
            const bf16* ap = &As[cur][(mt * 16 + l16) * LDK + half * 8];
            v8bf a0 = *(const v8bf*)(ap);
            v8bf a1 = *(const v8bf*)(ap + 16);
            #pragma unroll
            for (int i = 0; i < 8; ++i) { afrag[mt][i] = a0[i]; afrag[mt][i + 8] = a1[i]; }
        }
        __builtin_amdgcn_sched_barrier(0);   // loads above, WMMA chain below

        // ---- 8 back-to-back WMMAs (keep the matrix pipe saturated) -------
        #pragma unroll
        for (int mt = 0; mt < 8; ++mt) {
            acc[mt] = __builtin_amdgcn_wmma_f32_16x16x32_bf16(
                false, afrag[mt], false, bfrag, (short)0, acc[mt], false, false);
        }

        if (it + 1 < NK) {
            if (issuer) __builtin_amdgcn_s_wait_tensorcnt(0);
            __syncthreads();    // next buffer DMA'd; this buffer free to overwrite
        }
    }

    // ---- fused LIF epilogue ----------------------------------------------
    // C/D layout: VGPR r, lanes 0-15 -> M = tile*16 + r, lanes 16-31 -> +8.
    const int   n  = nBlock + wave * 16 + l16;
    const float bv = bias[n];
    #pragma unroll
    for (int mt = 0; mt < 8; ++mt) {
        #pragma unroll
        for (int r = 0; r < 8; ++r) {
            const int    m   = mt * 16 + half * 8 + r;
            const size_t idx = (size_t)m * N_TOTAL + n;
            float cur = acc[mt][r] + bv;
            float mv  = BETA * mem[idx] + cur;
            float s   = (mv > THR) ? 1.0f : 0.0f;
            mv       -= s * THR;
            mem[idx]  = mv;
            if (OUT_BF16) spk_bf16[idx] = (bf16)s;
            else          spk_f32[idx]  = s;
        }
    }
}

// ---------------------------------------------------------------------------
extern "C" void kernel_launch(void* const* d_in, const int* in_sizes, int n_in,
                              void* d_out, int out_size, void* d_ws, size_t ws_size,
                              hipStream_t stream)
{
    const float* x   = (const float*)d_in[0];   // [T][128][2048]
    const float* W1  = (const float*)d_in[1];   // [4096][2048]
    const float* b1  = (const float*)d_in[2];   // [4096]
    const float* W2  = (const float*)d_in[3];   // [1024][4096]
    const float* b2  = (const float*)d_in[4];   // [1024]
    float*       out = (float*)d_out;           // [T][128][1024]

    const int T = in_sizes[0] / (BATCH * IN_DIM);

    char* ws   = (char*)d_ws;
    bf16*  W1b  = (bf16*)(ws + WS_OFF_W1B);
    bf16*  W2b  = (bf16*)(ws + WS_OFF_W2B);
    float* mem1 = (float*)(ws + WS_OFF_MEM1);
    float* mem2 = (float*)(ws + WS_OFF_MEM2);
    bf16*  spk1 = (bf16*)(ws + WS_OFF_SPK1);

    // persistent LIF state starts at zero each launch (deterministic)
    hipMemsetAsync(mem1, 0, (size_t)BATCH * HID_DIM * sizeof(float), stream);
    hipMemsetAsync(mem2, 0, (size_t)BATCH * OUT_DIM * sizeof(float), stream);

    // one-shot bf16 weight conversion (stays L2-resident for all T steps)
    {
        int n1 = HID_DIM * IN_DIM;
        int n2 = OUT_DIM * HID_DIM;
        f32_to_bf16_kernel<<<(n1 / 4 + 255) / 256, 256, 0, stream>>>(W1, W1b, n1);
        f32_to_bf16_kernel<<<(n2 / 4 + 255) / 256, 256, 0, stream>>>(W2, W2b, n2);
    }

    for (int t = 0; t < T; ++t) {
        const float* xt   = x + (size_t)t * BATCH * IN_DIM;
        float*       outt = out + (size_t)t * BATCH * OUT_DIM;

        lif_gemm_kernel<HID_DIM, IN_DIM, true, true>
            <<<HID_DIM / 128, 256, 0, stream>>>(
                (const void*)xt, W1b, b1, mem1, spk1, nullptr);

        lif_gemm_kernel<OUT_DIM, HID_DIM, false, false>
            <<<OUT_DIM / 128, 256, 0, stream>>>(
                (const void*)spk1, W2b, b2, mem2, nullptr, outt);
    }
}
